// DeformableConv2d_76244259439171
// MI455X (gfx1250) — matrix operand order repaired and checked by reading, not compiled, not run on hardware
//
#include <hip/hip_runtime.h>
#include <hip/hip_bf16.h>

// ---------------------------------------------------------------------------
// DCNv2 forward for MI455X (gfx1250, wave32, WMMA).
//  kernel 0: pack fp32 weights -> f16, K-order kk = k*64 + c
//  kernel 1: offset+mask conv as WMMA GEMM (M=BHW, N=32, K=576), LDS weights
//  kernel 2: deformable sampling fused with WMMA GEMM (M=BHW, N=64, K=576)
// All gathers are unconditional (clamped addresses, validity folded into
// multipliers) so loads pipeline instead of exec-diverging.
// ---------------------------------------------------------------------------

typedef __attribute__((ext_vector_type(16))) _Float16 v16h;
typedef __attribute__((ext_vector_type(8)))  float    v8f;

#define Bn   4
#define Cc   64
#define Hh   128
#define Ww   128
#define HWs  (Hh * Ww)
#define K2c  9
#define Kdim (Cc * K2c)      /* 576 */
#define COUT 64
#define NTILES (Bn * HWs / 16)   /* 4096 M-tiles of 16 pixels */

union V16U { v16h h; uint4 u[2]; };

static __device__ inline v16h load_v16(const void* p) {
    V16U r;
    r.u[0] = reinterpret_cast<const uint4*>(p)[0];
    r.u[1] = reinterpret_cast<const uint4*>(p)[1];
    return r.h;
}

// ---------------------------------------------------------------------------
// Kernel 0: weight packing fp32 -> f16.
//   Wp [64][576] : main conv weights, kk = k*64 + c
//   Wom[32][576] : rows 0..17 = w_off, 18..26 = w_mask, 27..31 = 0
// ---------------------------------------------------------------------------
__global__ void dcn_pack_weights(const float* __restrict__ w,
                                 const float* __restrict__ w_off,
                                 const float* __restrict__ w_mask,
                                 _Float16* __restrict__ Wp,
                                 _Float16* __restrict__ Wom) {
    int idx = blockIdx.x * blockDim.x + threadIdx.x;
    if (idx < COUT * Kdim) {
        int o = idx / Kdim, kk = idx % Kdim;
        int k = kk >> 6, c = kk & 63;
        Wp[idx] = (_Float16)w[(o * Cc + c) * K2c + k];
    } else {
        int j = idx - COUT * Kdim;
        if (j < 32 * Kdim) {
            int n = j / Kdim, kk = j % Kdim;
            int k = kk >> 6, c = kk & 63;
            float v = 0.f;
            if (n < 18)      v = w_off[(n * Cc + c) * K2c + k];
            else if (n < 27) v = w_mask[((n - 18) * Cc + c) * K2c + k];
            Wom[j] = (_Float16)v;
        }
    }
}

// ---------------------------------------------------------------------------
// Kernel 1: offset + mask conv.  Per wave: one 16-pixel M-tile, N = 32.
// A fragment built in registers from the zero-padded 3x3 im2col of x.
// Padding handled by clamped (always-valid) addresses * {0,1} multiplier.
// ---------------------------------------------------------------------------
__global__ void __launch_bounds__(256)
dcn_offmask(const float* __restrict__ x,
            const float* __restrict__ b_off,
            const float* __restrict__ b_mask,
            const _Float16* __restrict__ Wom,
            float* __restrict__ off_ws,
            float* __restrict__ mask_ws) {
    __shared__ __align__(16) unsigned short wlds[32 * Kdim];   // 36 KB
    {   // cooperative global -> LDS weight stage (2304 x uint4)
        const uint4* src = reinterpret_cast<const uint4*>(Wom);
        uint4*       dst = reinterpret_cast<uint4*>(wlds);
        for (int i = threadIdx.x; i < (32 * Kdim) / 8; i += blockDim.x)
            dst[i] = src[i];
    }
    __syncthreads();

    const int wave = threadIdx.x >> 5;
    const int lane = threadIdx.x & 31;
    const int tile = blockIdx.x * 8 + wave;
    const int p0   = tile << 4;
    const int b    = p0 >> 14;
    const int rem  = p0 & 16383;
    const int h    = rem >> 7;
    const int w0   = rem & 127;
    const int wl   = w0 + (lane & 15);     // this lane's A-row pixel (w coord)
    const int hi   = lane >> 4;
    const float* xb = x + b * Cc * HWs;

    v8f acc0 = {}, acc1 = {};

    for (int k = 0; k < K2c; ++k) {
        const int ki = k / 3, kj = k % 3;
        const int yy = h - 1 + ki;
        const int xx = wl - 1 + kj;
        const float vf = ((yy >= 0) & (yy < Hh) & (xx >= 0) & (xx < Ww)) ? 1.f : 0.f;
        const int yc = min(max(yy, 0), Hh - 1);
        const int xc = min(max(xx, 0), Ww - 1);
        const float* xr = xb + yc * Ww + xc;   // always in-bounds

        for (int half = 0; half < 2; ++half) {
            const int cb = half * 32 + hi * 16;      // A-layout channel base
            // phase 1: issue all 16 gathers (independent, no divergence)
            float t[16];
            #pragma unroll
            for (int i = 0; i < 8; ++i) {
                t[i]     = xr[(cb + i) * HWs];
                t[8 + i] = xr[(cb + 8 + i) * HWs];
            }
            // phase 2: scale + convert
            v16h a;
            #pragma unroll
            for (int i = 0; i < 16; ++i)
                a[i] = (_Float16)(vf * t[i]);

            const int s    = 2 * k + half;
            const int koff = hi * 16;
            v16h bA = load_v16(&wlds[(      (lane & 15)) * Kdim + s * 32 + koff]);
            v16h bB = load_v16(&wlds[(16 + (lane & 15)) * Kdim + s * 32 + koff]);
            acc0 = __builtin_amdgcn_wmma_f32_16x16x32_f16(false, a, false, bA,
                                                          (short)0, acc0, false, false);
            acc1 = __builtin_amdgcn_wmma_f32_16x16x32_f16(false, a, false, bB,
                                                          (short)0, acc1, false, false);
        }
    }

    const int wbase = w0 + hi * 8;     // D rows m = r + 8*hi are consecutive w
    #pragma unroll
    for (int t = 0; t < 2; ++t) {
        const int o = t * 16 + (lane & 15);
        v8f acc = t ? acc1 : acc0;
        if (o < 18) {
            const float bias = b_off[o];
            float* dst = off_ws + ((b * 18 + o) * Hh + h) * Ww + wbase;
            #pragma unroll
            for (int r = 0; r < 8; ++r) dst[r] = acc[r] + bias;
        } else if (o < 27) {
            const float bias = b_mask[o - 18];
            float* dst = mask_ws + ((b * 9 + (o - 18)) * Hh + h) * Ww + wbase;
            #pragma unroll
            for (int r = 0; r < 8; ++r) {
                float v = acc[r] + bias;
                dst[r] = 1.f / (1.f + __expf(-v));
            }
        }
    }
}

// ---------------------------------------------------------------------------
// Kernel 2: deformable bilinear sampling fused with the main WMMA GEMM.
// Per wave: one 16-pixel M-tile x 4 N-tiles (Cout=64). A fragment is the
// bilinear-sampled, mask-scaled activation built directly in registers.
// Gathers issued per-corner into temp arrays so 32 loads stay in flight.
// ---------------------------------------------------------------------------
__global__ void __launch_bounds__(256)
dcn_main(const float* __restrict__ x,
         const float* __restrict__ bvec,
         const _Float16* __restrict__ Wp,
         const float* __restrict__ off_ws,
         const float* __restrict__ mask_ws,
         float* __restrict__ out) {
    const int wave = threadIdx.x >> 5;
    const int lane = threadIdx.x & 31;
    const int tile = blockIdx.x * 8 + wave;
    const int p0   = tile << 4;
    const int b    = p0 >> 14;
    const int rem  = p0 & 16383;
    const int h    = rem >> 7;
    const int w0   = rem & 127;
    const int wl   = w0 + (lane & 15);
    const int hi   = lane >> 4;
    const float* xb = x + b * Cc * HWs;

    // pull the 72 KB weight panel toward L2/WGP$ early
    __builtin_prefetch(Wp + lane * 64, 0, 3);

    v8f acc[4] = {};

    for (int k = 0; k < K2c; ++k) {
        const int ki = k / 3, kj = k % 3;
        const int po = (b * 18 + 2 * k) * HWs + h * Ww + wl;
        const float dy   = off_ws[po];
        const float dx   = off_ws[po + HWs];
        const float mval = mask_ws[(b * 9 + k) * HWs + h * Ww + wl];

        const float sy  = (float)(h - 1 + ki) + dy;
        const float sx  = (float)(wl - 1 + kj) + dx;
        const float y0f = floorf(sy), x0f = floorf(sx);
        const int   y0  = (int)y0f,   x0i = (int)x0f;
        const float wy1 = sy - y0f,   wx1 = sx - x0f;
        const float wy0 = 1.f - wy1,  wx0 = 1.f - wx1;

        int   idxs[4];
        float wgts[4];
        #pragma unroll
        for (int cr = 0; cr < 4; ++cr) {
            const int yi = y0 + (cr >> 1), xi = x0i + (cr & 1);
            const bool v = (yi >= 0) & (yi < Hh) & (xi >= 0) & (xi < Ww);
            const int yc = min(max(yi, 0), Hh - 1);
            const int xc = min(max(xi, 0), Ww - 1);
            idxs[cr] = yc * Ww + xc;
            const float wy = (cr >> 1) ? wy1 : wy0;
            const float wx = (cr & 1)  ? wx1 : wx0;
            wgts[cr] = v ? wy * wx * mval : 0.f;   // fold validity + mask
        }

        for (int half = 0; half < 2; ++half) {
            const int cb = half * 32 + hi * 16;
            v16h a;
            #pragma unroll
            for (int g = 0; g < 2; ++g) {
                const int cg = cb + g * 8;
                // phase 1: 32 independent gathers (4 corners x 8 channels)
                float v0[8], v1[8], v2[8], v3[8];
                #pragma unroll
                for (int i = 0; i < 8; ++i) {
                    const float* pc = xb + (cg + i) * HWs;
                    v0[i] = pc[idxs[0]];
                    v1[i] = pc[idxs[1]];
                    v2[i] = pc[idxs[2]];
                    v3[i] = pc[idxs[3]];
                }
                // phase 2: bilinear combine + convert
                #pragma unroll
                for (int i = 0; i < 8; ++i) {
                    float v = wgts[0] * v0[i] + wgts[1] * v1[i]
                            + wgts[2] * v2[i] + wgts[3] * v3[i];
                    a[g * 8 + i] = (_Float16)v;
                }
            }
            const int s    = 2 * k + half;
            const int koff = hi * 16;
            #pragma unroll
            for (int t = 0; t < 4; ++t) {
                v16h bf = load_v16(&Wp[(t * 16 + (lane & 15)) * Kdim + s * 32 + koff]);
                acc[t] = __builtin_amdgcn_wmma_f32_16x16x32_f16(false, a, false, bf,
                                                                (short)0, acc[t],
                                                                false, false);
            }
        }
    }

    const int wbase = w0 + hi * 8;
    #pragma unroll
    for (int t = 0; t < 4; ++t) {
        const int o  = t * 16 + (lane & 15);
        const float bo = bvec[o];
        float* dst = out + ((b * COUT + o) * Hh + h) * Ww + wbase;
        float4 lo = make_float4(acc[t][0] + bo, acc[t][1] + bo,
                                acc[t][2] + bo, acc[t][3] + bo);
        float4 hi4 = make_float4(acc[t][4] + bo, acc[t][5] + bo,
                                 acc[t][6] + bo, acc[t][7] + bo);
        *reinterpret_cast<float4*>(dst)     = lo;
        *reinterpret_cast<float4*>(dst + 4) = hi4;
    }
}

// ---------------------------------------------------------------------------
extern "C" void kernel_launch(void* const* d_in, const int* in_sizes, int n_in,
                              void* d_out, int out_size, void* d_ws, size_t ws_size,
                              hipStream_t stream) {
    const float* x      = (const float*)d_in[0];
    const float* w_off  = (const float*)d_in[1];
    const float* b_off  = (const float*)d_in[2];
    const float* w_mask = (const float*)d_in[3];
    const float* b_mask = (const float*)d_in[4];
    const float* w      = (const float*)d_in[5];
    const float* bvec   = (const float*)d_in[6];
    float* out = (float*)d_out;

    char* ws = (char*)d_ws;
    _Float16* Wp   = (_Float16*)ws;                          // 73728 B
    _Float16* Wom  = (_Float16*)(ws + 73728);                // 36864 B
    float* off_ws  = (float*)(ws + 73728 + 36864);           // 4*18*HW fp32
    float* mask_ws = off_ws + (size_t)Bn * 18 * HWs;         // 4*9*HW fp32

    // kernel 0: pack weights (36864 + 18432 = 55296 elements)
    dcn_pack_weights<<<216, 256, 0, stream>>>(w, w_off, w_mask, Wp, Wom);
    // kernel 1: offset/mask conv, 4096 M-tiles, 8 waves per block
    dcn_offmask<<<NTILES / 8, 256, 0, stream>>>(x, b_off, b_mask, Wom,
                                                off_ws, mask_ws);
    // kernel 2: fused deformable sampling + GEMM
    dcn_main<<<NTILES / 8, 256, 0, stream>>>(x, bvec, Wp, off_ws, mask_ws, out);
}
